// BinConv2d_58695023067837
// MI455X (gfx1250) — 8-bit matrix operand order repaired and checked by reading, compiled, not run on hardware
//
#include <hip/hip_runtime.h>
#include <stdint.h>

typedef __attribute__((ext_vector_type(16))) int   v16i;
typedef __attribute__((ext_vector_type(8)))  float v8f;

#define CIN    256
#define COUT   256
#define HH     56
#define WW     56
#define HW     (HH * WW)
#define BATCH  32
#define NPIX   (BATCH * HW)
#define KTOT   (9 * CIN)

#define MTILE  64
#define NTILE  128
#define A_STRIDE 272
#define B_STRIDE 272

#define FP8_P1 0x38u
#define FP8_M1 0xB8u

__global__ void bn_sign_pack_kernel(const float* __restrict__ x,
                                    const float* __restrict__ gamma,
                                    const float* __restrict__ beta,
                                    const float* __restrict__ rmean,
                                    const float* __restrict__ rvar,
                                    uint8_t* __restrict__ xq) {
    int tid   = blockIdx.x * blockDim.x + threadIdx.x;
    int pixel = tid >> 6;
    int c0    = (tid & 63) << 2;
    if (pixel >= NPIX) return;
    int b   = pixel / HW;
    int rem = pixel - b * HW;
    const float* xp = x + (size_t)b * (CIN * HW) + rem;
    uint32_t packed = 0;
#pragma unroll
    for (int i = 0; i < 4; ++i) {
        int c = c0 + i;
        float inv  = gamma[c] * rsqrtf(rvar[c] + 1e-4f);
        float v    = xp[(size_t)c * HW] * inv + (beta[c] - rmean[c] * inv);
        uint32_t s = (v >= 0.0f) ? FP8_P1 : FP8_M1;
        packed |= s << (8 * i);
    }
    *(uint32_t*)(xq + (size_t)pixel * CIN + c0) = packed;
}

__global__ void wsign_pack_kernel(const float* __restrict__ w,
                                  uint8_t* __restrict__ wq,
                                  float* __restrict__ scale) {
    __shared__ float red[256];
    int o  = blockIdx.x;
    int ci = threadIdx.x;
    const float* wp = w + ((size_t)o * CIN + ci) * 9;
    float s = 0.0f;
#pragma unroll
    for (int t = 0; t < 9; ++t) {
        float v = wp[t];
        s += fabsf(v);
        wq[(size_t)o * KTOT + t * CIN + ci] = (v >= 0.0f) ? (uint8_t)FP8_P1
                                                          : (uint8_t)FP8_M1;
    }
    red[ci] = s;
    __syncthreads();
    for (int off = 128; off > 0; off >>= 1) {
        if (ci < off) red[ci] += red[ci + off];
        __syncthreads();
    }
    if (ci == 0) scale[o] = red[0] * (1.0f / (float)KTOT);
}

__device__ __forceinline__ void async_copy_b128(void* lds_dst, const void* gsrc) {
    uint32_t lds_off = (uint32_t)(uintptr_t)lds_dst;
    uint64_t gaddr   = (uint64_t)(uintptr_t)gsrc;
    asm volatile("global_load_async_to_lds_b128 %0, %1, off"
                 :: "v"(lds_off), "v"(gaddr) : "memory");
}
__device__ __forceinline__ void async_wait0() {
    asm volatile("s_wait_asynccnt 0x0" ::: "memory");
}

__device__ __forceinline__ void stage_tap(int tap,
                                          uint8_t* __restrict__ Ab,
                                          uint8_t* __restrict__ Bb,
                                          const uint8_t* __restrict__ xq,
                                          const uint8_t* __restrict__ wq,
                                          int n_base, int br, int bh,
                                          int sp, int sq, int sb, int sy, int sx) {
    {
        const uint8_t* src = wq + (size_t)(n_base + br) * KTOT + tap * CIN + bh * 128;
        uint8_t*       dst = Bb + br * B_STRIDE + bh * 128;
#pragma unroll
        for (int i = 0; i < 8; ++i)
            async_copy_b128(dst + i * 16, src + i * 16);
        if (tap < 8)
            __builtin_prefetch(wq + (size_t)(n_base + br) * KTOT +
                               (tap + 1) * CIN + bh * 128, 0, 0);
    }
    {
        const int ky = tap / 3;
        const int kx = tap - ky * 3;
        const int iy = sy + ky - 1;
        const int ix = sx + kx - 1;
        uint8_t* dst = Ab + sp * A_STRIDE + sq * 64;
        if (iy >= 0 && iy < HH && ix >= 0 && ix < WW) {
            const uint8_t* src =
                xq + ((size_t)(sb * HH + iy) * WW + ix) * CIN + sq * 64;
#pragma unroll
            for (int i = 0; i < 4; ++i)
                async_copy_b128(dst + i * 16, src + i * 16);
        } else {
            const int4 z = {0, 0, 0, 0};
#pragma unroll
            for (int i = 0; i < 4; ++i)
                *(int4*)(dst + i * 16) = z;
        }
    }
}

__global__ __launch_bounds__(256)
void binconv_wmma_kernel(const uint8_t* __restrict__ xq,
                         const uint8_t* __restrict__ wq,
                         const float*   __restrict__ scale,
                         float*         __restrict__ out) {
    __shared__ __align__(16) uint8_t As[2][MTILE * A_STRIDE];
    __shared__ __align__(16) uint8_t Bs[2][NTILE * B_STRIDE];

    const int tid  = threadIdx.x;
    const int lane = tid & 31;
    const int wave = tid >> 5;
    const int wm   = wave & 3;
    const int wn   = wave >> 2;
    const int half = lane >> 4;
    const int ln16 = lane & 15;

    const int m_base = blockIdx.x * MTILE;
    const int n_base = blockIdx.y * NTILE;

    v8f acc[4];
    const v8f zero = {0.f, 0.f, 0.f, 0.f, 0.f, 0.f, 0.f, 0.f};
#pragma unroll
    for (int i = 0; i < 4; ++i) acc[i] = zero;

    const int sp   = tid >> 2;
    const int sq   = tid & 3;
    const int sid  = m_base + sp;
    const int sb   = sid / HW;
    const int srem = sid - sb * HW;
    const int sy   = srem / WW;
    const int sx   = srem - sy * WW;
    const int br   = tid >> 1;
    const int bh   = tid & 1;

    stage_tap(0, As[0], Bs[0], xq, wq, n_base, br, bh, sp, sq, sb, sy, sx);
    async_wait0();
    __syncthreads();

    for (int tap = 0; tap < 9; ++tap) {
        const int cur = tap & 1;

        if (tap + 1 < 9)
            stage_tap(tap + 1, As[cur ^ 1], Bs[cur ^ 1],
                      xq, wq, n_base, br, bh, sp, sq, sb, sy, sx);

        const uint8_t* Abuf = As[cur];
        const uint8_t* Bbuf = Bs[cur];
#pragma unroll
        for (int ks = 0; ks < 2; ++ks) {
            const uint8_t* ap =
                Abuf + (wm * 16 + ln16) * A_STRIDE + ks * 128 + half * 16;
            union { v16i v; int2 d[8]; } au;
#pragma unroll
            for (int j = 0; j < 8; ++j)
                au.d[j] = *(const int2*)(ap + (j >> 1) * 32 + (j & 1) * 8);
#pragma unroll
            for (int nf = 0; nf < 4; ++nf) {
                const uint8_t* bp =
                    Bbuf + (wn * 64 + nf * 16 + ln16) * B_STRIDE + ks * 128 + half * 16;
                union { v16i v; int4 q[4]; } bu;
                bu.q[0] = *(const int4*)(bp + 0);
                bu.q[1] = *(const int4*)(bp + 32);
                bu.q[2] = *(const int4*)(bp + 64);
                bu.q[3] = *(const int4*)(bp + 96);
                acc[nf] = __builtin_amdgcn_wmma_f32_16x16x128_fp8_fp8(
                    au.v, bu.v, (short)0, acc[nf],
                     false,  false);
            }
        }

        if (tap + 1 < 9) {
            async_wait0();
            __syncthreads();
        }
    }

    int obase[8];
#pragma unroll
    for (int r = 0; r < 8; ++r) {
        int id  = m_base + wm * 16 + half * 8 + r;
        int bb  = id / HW;
        int rem = id - bb * HW;
        obase[r] = bb * (COUT * HW) + rem;
    }
#pragma unroll
    for (int nf = 0; nf < 4; ++nf) {
        int   o  = n_base + wn * 64 + nf * 16 + ln16;
        float sc = scale[o];
#pragma unroll
        for (int r = 0; r < 8; ++r)
            out[(size_t)obase[r] + (size_t)o * HW] = sc * acc[nf][r];
    }
}

extern "C" void kernel_launch(void* const* d_in, const int* in_sizes, int n_in,
                              void* d_out, int out_size, void* d_ws, size_t ws_size,
                              hipStream_t stream) {
    const float* x     = (const float*)d_in[0];
    const float* gamma = (const float*)d_in[1];
    const float* beta  = (const float*)d_in[2];
    const float* rmean = (const float*)d_in[3];
    const float* rvar  = (const float*)d_in[4];
    const float* w     = (const float*)d_in[5];
    float* out = (float*)d_out;

    uint8_t* xq    = (uint8_t*)d_ws;
    uint8_t* wq    = (uint8_t*)d_ws + (size_t)NPIX * CIN;
    float*   scale = (float*)((char*)d_ws + (size_t)NPIX * CIN
                                           + (size_t)COUT * KTOT);

    bn_sign_pack_kernel<<<(NPIX * 64) / 256, 256, 0, stream>>>(
        x, gamma, beta, rmean, rvar, xq);

    wsign_pack_kernel<<<COUT, 256, 0, stream>>>(w, wq, scale);

    dim3 grid(NPIX / MTILE, COUT / NTILE);
    binconv_wmma_kernel<<<grid, 256, 0, stream>>>(xq, wq, scale, out);
}
